// LlamaAttention_52536039965364
// MI455X (gfx1250) — matrix operand order repaired and checked by reading, not compiled, not run on hardware
//
#include <hip/hip_runtime.h>
#include <hip/hip_bf16.h>

// ---------------------------------------------------------------------------
// LLaMA attention block on gfx1250 (MI455X): bf16 WMMA + TDM tile staging.
//   HIDDEN=2048, NH=32, NKV=8, HD=64, G=4, B=2, S=2048
// ---------------------------------------------------------------------------

#define HID   2048
#define NH    32
#define NKV   8
#define HD    64
#define SEQ   2048
#define BSZ   2
#define TOKS  (BSZ * SEQ)        // 4096

typedef __attribute__((ext_vector_type(8)))  __bf16 v8bf;
typedef __attribute__((ext_vector_type(16))) __bf16 v16bf;
typedef __attribute__((ext_vector_type(8)))  float  v8f;

#if __has_builtin(__builtin_amdgcn_tensor_load_to_lds) && \
    __has_builtin(__builtin_amdgcn_s_wait_tensorcnt)
#define USE_TDM 1
#else
#define USE_TDM 0
#endif

#if __has_builtin(__builtin_amdgcn_update_dpp)
#define USE_DPP 1
#else
#define USE_DPP 0
#endif

static __device__ __forceinline__ __bf16 f2bf(float f) {
  unsigned u = __float_as_uint(f);
  u += 0x7FFFu + ((u >> 16) & 1u);              // round-to-nearest-even
  unsigned short h = (unsigned short)(u >> 16);
  return __builtin_bit_cast(__bf16, h);
}
static __device__ __forceinline__ float bf2f(__bf16 b) {
  unsigned short h = __builtin_bit_cast(unsigned short, b);
  return __uint_as_float(((unsigned)h) << 16);
}
static __device__ __forceinline__ v16bf cat16(v8bf lo, v8bf hi) {
  return __builtin_shufflevector(lo, hi, 0,1,2,3,4,5,6,7,8,9,10,11,12,13,14,15);
}

// ---- 16-lane (half-wave) butterfly reductions --------------------------------
// DPP path: quad_perm(xor1), quad_perm(xor2), row_half_mirror, row_mirror
// form a complete reduction across each group of 16 lanes (wave32 rows).
#if USE_DPP
template <int CTRL>
static __device__ __forceinline__ float dppf(float x) {
  int i = __float_as_int(x);
  i = __builtin_amdgcn_update_dpp(0, i, CTRL, 0xF, 0xF, true);
  return __int_as_float(i);
}
static __device__ __forceinline__ float rmax16(float v) {
  v = fmaxf(v, dppf<0xB1>(v));    // quad_perm [1,0,3,2]  (xor 1)
  v = fmaxf(v, dppf<0x4E>(v));    // quad_perm [2,3,0,1]  (xor 2)
  v = fmaxf(v, dppf<0x141>(v));   // row_half_mirror      (xor 4 at quad level)
  v = fmaxf(v, dppf<0x140>(v));   // row_mirror           (xor 8 at octet level)
  return v;
}
static __device__ __forceinline__ float rsum16(float v) {
  v += dppf<0xB1>(v);
  v += dppf<0x4E>(v);
  v += dppf<0x141>(v);
  v += dppf<0x140>(v);
  return v;
}
#else
static __device__ __forceinline__ float rmax16(float v) {
#pragma unroll
  for (int m = 1; m < 16; m <<= 1) v = fmaxf(v, __shfl_xor(v, m, 32));
  return v;
}
static __device__ __forceinline__ float rsum16(float v) {
#pragma unroll
  for (int m = 1; m < 16; m <<= 1) v += __shfl_xor(v, m, 32);
  return v;
}
#endif

// ---- Tensor Data Mover: 2D tile (bf16) global -> LDS -------------------------
#if USE_TDM
typedef __attribute__((ext_vector_type(4))) unsigned int tdm_g0_t;
typedef __attribute__((ext_vector_type(8))) int          tdm_g1_t;
typedef __attribute__((ext_vector_type(4))) int          tdm_g2_t;

// Generic LDS pointers carry the LDS byte offset in addr[31:0] (flat aperture).
static __device__ __forceinline__ unsigned lds_lo32(const void* p) {
  return (unsigned)(unsigned long long)(size_t)p;
}

// D# per CDNA5 ISA: group0 = {count/lds_addr/global_addr,
// type=2}, group1 = {data_size=2B, tensor_dim0/1, tile_dim0/1, dim0 stride}.
static __device__ __forceinline__ void tdm_load_2d(
    unsigned lds_byte_addr, const void* gaddr,
    unsigned tile_d0, unsigned tile_d1,                  // tile, elements
    unsigned tensor_d0, unsigned tensor_d1,              // tensor, elements
    unsigned long long stride_d0)                        // row stride, elements
{
  unsigned long long ga = (unsigned long long)(size_t)gaddr;
  tdm_g0_t g0;
  g0[0] = 1u;                                            // count=1, user D#
  g0[1] = lds_byte_addr;                                 // lds_addr
  g0[2] = (unsigned)(ga & 0xFFFFFFFFu);                  // global_addr[31:0]
  g0[3] = (unsigned)((ga >> 32) & 0x01FFFFFFu)           // global_addr[56:32]
          | 0x80000000u;                                 // type=2 ("image")
  tdm_g1_t g1;
  g1[0] = (int)(1u << 16);                               // data_size=1 (2 bytes)
  g1[1] = (int)(tensor_d0 << 16);                        // tensor_dim0[15:0]
  g1[2] = (int)((tensor_d0 >> 16) | (tensor_d1 << 16));  // dim0[31:16]|dim1[15:0]
  g1[3] = (int)((tensor_d1 >> 16) | (tile_d0 << 16));    // dim1[31:16]|tile_dim0
  g1[4] = (int)(tile_d1 & 0xFFFFu);                      // tile_dim1 (tile_dim2=0)
  g1[5] = (int)(stride_d0 & 0xFFFFFFFFull);              // dim0_stride[31:0]
  g1[6] = (int)((stride_d0 >> 32) & 0xFFFFull);          // dim0_stride[47:32]
  g1[7] = 0;
  tdm_g2_t gz  = {0, 0, 0, 0};
  tdm_g1_t gz8 = {0, 0, 0, 0, 0, 0, 0, 0};
  __builtin_amdgcn_tensor_load_to_lds(g0, g1, gz, gz, gz8, 0);
}
#endif

// ---------------------------------------------------------------------------
// fp32 -> bf16 conversion
// ---------------------------------------------------------------------------
__global__ void cvt_f32_to_bf16(const float* __restrict__ src,
                                __bf16* __restrict__ dst, int n) {
  int i = blockIdx.x * blockDim.x + threadIdx.x;
  if (i < n) dst[i] = f2bf(src[i]);
}

// ---------------------------------------------------------------------------
// Tiled WMMA GEMM: C[M,N] = A[M,K] * B[K,N], A/B bf16 row-major.
// Block = 256 threads (8 waves), tile 128(M) x 64(N), BK = 32.
// A tile staged by TDM (wave 0 issues one 2D descriptor, s_wait_tensorcnt);
// B tile staged transposed so B fragments are contiguous ds_load_b128s.
// ---------------------------------------------------------------------------
template <bool OUT_BF16>
__global__ __launch_bounds__(256)
void gemm_wmma(const __bf16* __restrict__ A, const __bf16* __restrict__ B,
               void* __restrict__ Cout, int M, int N, int K) {
  __shared__ __align__(16) __bf16 As[128][32];
  __shared__ __align__(16) __bf16 Bt[64][32];

  const int tid  = threadIdx.x;
  const int wave = tid >> 5, lane = tid & 31;
  const int l15  = lane & 15, hlf = lane >> 4;
  const int wr   = wave >> 1, wc = wave & 1;
  const int m0   = blockIdx.y * 128;
  const int n0   = blockIdx.x * 64;

  v8f c[2][2] = {};

  for (int k0 = 0; k0 < K; k0 += 32) {
    __syncthreads();
#if USE_TDM
    if (wave == 0) {
      tdm_load_2d(lds_lo32(&As[0][0]), &A[(size_t)m0 * K + k0],
                  /*tile*/ 32, 128, /*tensor*/ (unsigned)K, (unsigned)M,
                  (unsigned long long)K);
      __builtin_amdgcn_s_wait_tensorcnt(0);
    }
#else
#pragma unroll
    for (int i = 0; i < 2; ++i) {
      int v = tid + i * 256;              // 0..511
      int row = v >> 2, ko = (v & 3) * 8;
      *(v8bf*)&As[row][ko] =
          *(const v8bf*)&A[(size_t)(m0 + row) * K + k0 + ko];
    }
#endif
    // Stage B tile 32x64 transposed into Bt[64][32]
    {
      int kk = tid >> 3, no = (tid & 7) * 8;
      v8bf b = *(const v8bf*)&B[(size_t)(k0 + kk) * N + n0 + no];
      if (k0 + 32 < K)
        __builtin_prefetch(&B[(size_t)(k0 + 32 + kk) * N + n0 + no], 0, 1);
#pragma unroll
      for (int j = 0; j < 8; ++j) Bt[no + j][kk] = b[j];
    }
    __syncthreads();

    v16bf a[2], bf[2];
#pragma unroll
    for (int mi = 0; mi < 2; ++mi) {
      int row = wr * 32 + mi * 16 + l15;
      a[mi] = cat16(*(const v8bf*)&As[row][hlf * 8],
                    *(const v8bf*)&As[row][16 + hlf * 8]);
    }
#pragma unroll
    for (int ni = 0; ni < 2; ++ni) {
      int col = wc * 32 + ni * 16 + l15;
      bf[ni] = cat16(*(const v8bf*)&Bt[col][hlf * 8],
                     *(const v8bf*)&Bt[col][16 + hlf * 8]);
    }
#pragma unroll
    for (int mi = 0; mi < 2; ++mi)
#pragma unroll
      for (int ni = 0; ni < 2; ++ni)
        c[mi][ni] = __builtin_amdgcn_wmma_f32_16x16x32_bf16(
            false, a[mi], false, bf[ni], (short)0, c[mi][ni], false, false);
  }

#pragma unroll
  for (int mi = 0; mi < 2; ++mi)
#pragma unroll
    for (int ni = 0; ni < 2; ++ni)
#pragma unroll
      for (int r = 0; r < 8; ++r) {
        int row = m0 + wr * 32 + mi * 16 + r + 8 * hlf;
        int col = n0 + wc * 32 + ni * 16 + l15;
        float v = c[mi][ni][r];
        if (OUT_BF16)
          ((__bf16*)Cout)[(size_t)row * N + col] = f2bf(v);
        else
          ((float*)Cout)[(size_t)row * N + col] = v;
      }
}

// ---------------------------------------------------------------------------
// RoPE (in-place, bf16) with optional output scaling (folds 1/sqrt(HD) into Q).
// One thread per (token, head, i<32).
// ---------------------------------------------------------------------------
__global__ void rope_kernel(__bf16* __restrict__ data, int nheads, float scale) {
  int idx  = blockIdx.x * 256 + threadIdx.x;
  int i    = idx & 31;
  int rest = idx >> 5;
  int h    = rest % nheads;
  int tok  = rest / nheads;                 // 0..TOKS-1
  int pos  = tok & (SEQ - 1);
  __bf16* p = data + (size_t)tok * (nheads * HD) + h * HD;
  // inv_freq = 10000^(-2i/64) = exp(-2i/64 * ln(10000))
  float inv = __expf(-(float)(2 * i) * (9.210340371976184f / 64.0f));
  float ang = (float)pos * inv;
  float sn, cs;
  __sincosf(ang, &sn, &cs);
  float a = bf2f(p[i]), b = bf2f(p[i + 32]);
  p[i]      = f2bf((a * cs - b * sn) * scale);
  p[i + 32] = f2bf((b * cs + a * sn) * scale);
}

// ---------------------------------------------------------------------------
// Flash attention (GQA): block = 8 waves; each wave = 16 query rows of one
// (batch, q-head). K chunk staged via TDM; V chunk staged transposed.
// Scores + P*V via v_wmma_f32_16x16x32_bf16; online softmax in fp32 with
// DPP butterfly reductions across each 16-lane half.
// ---------------------------------------------------------------------------
__global__ __launch_bounds__(256)
void attn_wmma(const __bf16* __restrict__ Q, const __bf16* __restrict__ Kg,
               const __bf16* __restrict__ Vg, __bf16* __restrict__ O) {
  __shared__ __align__(16) __bf16 Ks[32][HD];     // keys chunk, row-major
  __shared__ __align__(16) __bf16 Vt[HD][32];     // V chunk transposed
  __shared__ __align__(16) __bf16 Pl[8][16][32];  // per-wave P relayout

  const int tid  = threadIdx.x;
  const int wave = tid >> 5, lane = tid & 31;
  const int l15  = lane & 15, hlf = lane >> 4;
  const int h    = blockIdx.y;                 // query head
  const int b    = blockIdx.z;
  const int kvh  = h >> 2;                     // GROUPS = 4
  const int qbase = (blockIdx.x * 8 + wave) * 16;
  const size_t tok0 = (size_t)b * SEQ;

  // Loop-invariant Q fragments (16 rows x 64 dims = two 16x32 A-frags)
  const __bf16* qrow = Q + (tok0 + qbase + l15) * (NH * HD) + h * HD;
  v16bf aq0 = cat16(*(const v8bf*)(qrow + hlf * 8),
                    *(const v8bf*)(qrow + 16 + hlf * 8));
  v16bf aq1 = cat16(*(const v8bf*)(qrow + 32 + hlf * 8),
                    *(const v8bf*)(qrow + 48 + hlf * 8));

  v8f o[4] = {};
  float m[8], l[8];
#pragma unroll
  for (int r = 0; r < 8; ++r) { m[r] = -1e30f; l[r] = 0.0f; }

  for (int t0 = 0; t0 < SEQ; t0 += 32) {
    __syncthreads();
    // Stage K chunk [32][64] via TDM; V chunk transposed by all threads.
#if USE_TDM
    if (wave == 0) {
      tdm_load_2d(lds_lo32(&Ks[0][0]),
                  &Kg[(tok0 + t0) * (NKV * HD) + kvh * HD],
                  /*tile*/ HD, 32, /*tensor*/ NKV * HD, TOKS,
                  (unsigned long long)(NKV * HD));
      __builtin_amdgcn_s_wait_tensorcnt(0);
    }
    {
      int row = tid >> 3, ko = (tid & 7) * 8;
      v8bf vv = *(const v8bf*)&Vg[(tok0 + t0 + row) * (NKV * HD) + kvh * HD + ko];
#pragma unroll
      for (int j = 0; j < 8; ++j) Vt[ko + j][row] = vv[j];
    }
#else
    {
      int row = tid >> 3, ko = (tid & 7) * 8;
      size_t base = (tok0 + t0 + row) * (NKV * HD) + kvh * HD + ko;
      *(v8bf*)&Ks[row][ko] = *(const v8bf*)&Kg[base];
      v8bf vv = *(const v8bf*)&Vg[base];
#pragma unroll
      for (int j = 0; j < 8; ++j) Vt[ko + j][row] = vv[j];
    }
#endif
    __syncthreads();

    // Scores: S[16q x 32keys] as two 16x16 C fragments (Q pre-scaled by 1/8)
    v8f s[2];
#pragma unroll
    for (int kh = 0; kh < 2; ++kh) {
      int token = kh * 16 + l15;
      v16bf b0 = cat16(*(const v8bf*)&Ks[token][hlf * 8],
                       *(const v8bf*)&Ks[token][16 + hlf * 8]);
      v16bf b1 = cat16(*(const v8bf*)&Ks[token][32 + hlf * 8],
                       *(const v8bf*)&Ks[token][48 + hlf * 8]);
      v8f acc = {};
      acc = __builtin_amdgcn_wmma_f32_16x16x32_bf16(false, aq0, false, b0,
                                                    (short)0, acc, false, false);
      acc = __builtin_amdgcn_wmma_f32_16x16x32_bf16(false, aq1, false, b1,
                                                    (short)0, acc, false, false);
      s[kh] = acc;
    }

    // Online softmax update (row = r + 8*hlf, col = lane&15 per C layout)
    float p0[8], p1[8], alpha[8];
#pragma unroll
    for (int r = 0; r < 8; ++r) {
      float a0 = s[0][r], a1 = s[1][r];
      float mx = rmax16(fmaxf(a0, a1));
      float mn = fmaxf(m[r], mx);
      alpha[r] = __expf(m[r] - mn);
      p0[r] = __expf(a0 - mn);
      p1[r] = __expf(a1 - mn);
      float rs = rsum16(p0[r] + p1[r]);
      l[r] = l[r] * alpha[r] + rs;
      m[r] = mn;
    }
#pragma unroll
    for (int d = 0; d < 4; ++d)
#pragma unroll
      for (int r = 0; r < 8; ++r) o[d][r] *= alpha[r];

    // Relayout P (C layout) -> A fragment via per-wave LDS buffer
#pragma unroll
    for (int r = 0; r < 8; ++r) {
      Pl[wave][r + 8 * hlf][l15]      = f2bf(p0[r]);
      Pl[wave][r + 8 * hlf][16 + l15] = f2bf(p1[r]);
    }
    __syncthreads();
    v16bf pA = cat16(*(const v8bf*)&Pl[wave][l15][hlf * 8],
                     *(const v8bf*)&Pl[wave][l15][16 + hlf * 8]);

    // O += P * V  (four 16x16 output fragments over head_dim)
#pragma unroll
    for (int d = 0; d < 4; ++d) {
      int dim = d * 16 + l15;
      v16bf bv = cat16(*(const v8bf*)&Vt[dim][hlf * 8],
                       *(const v8bf*)&Vt[dim][16 + hlf * 8]);
      o[d] = __builtin_amdgcn_wmma_f32_16x16x32_bf16(false, pA, false, bv,
                                                     (short)0, o[d], false, false);
    }
  }

  // Normalize and store attention output (bf16, [token][h*64+dim])
#pragma unroll
  for (int d = 0; d < 4; ++d)
#pragma unroll
    for (int r = 0; r < 8; ++r) {
      size_t row = tok0 + qbase + r + 8 * hlf;
      O[row * (NH * HD) + h * HD + d * 16 + l15] = f2bf(o[d][r] / l[r]);
    }
}

// ---------------------------------------------------------------------------
// Host driver
// ---------------------------------------------------------------------------
extern "C" void kernel_launch(void* const* d_in, const int* in_sizes, int n_in,
                              void* d_out, int out_size, void* d_ws,
                              size_t ws_size, hipStream_t stream) {
  const float* x  = (const float*)d_in[0];
  const float* wq = (const float*)d_in[1];
  const float* wk = (const float*)d_in[2];
  const float* wv = (const float*)d_in[3];
  const float* wo = (const float*)d_in[4];
  float* out = (float*)d_out;

  char* ws = (char*)d_ws;
  size_t off = 0;
  auto alloc = [&](size_t elems) {
    void* p = ws + off;
    off += (elems * sizeof(__bf16) + 255) & ~(size_t)255;
    return (__bf16*)p;
  };
  __bf16* xb  = alloc((size_t)TOKS * HID);
  __bf16* wqb = alloc((size_t)HID * (NH * HD));
  __bf16* wkb = alloc((size_t)HID * (NKV * HD));
  __bf16* wvb = alloc((size_t)HID * (NKV * HD));
  __bf16* wob = alloc((size_t)(NH * HD) * HID);
  __bf16* qb  = alloc((size_t)TOKS * (NH * HD));
  __bf16* kb  = alloc((size_t)TOKS * (NKV * HD));
  __bf16* vb  = alloc((size_t)TOKS * (NKV * HD));
  __bf16* ab  = alloc((size_t)TOKS * (NH * HD));

  auto cvt = [&](const float* s, __bf16* d, int n) {
    cvt_f32_to_bf16<<<(n + 255) / 256, 256, 0, stream>>>(s, d, n);
  };
  cvt(x,  xb,  TOKS * HID);
  cvt(wq, wqb, HID * NH * HD);
  cvt(wk, wkb, HID * NKV * HD);
  cvt(wv, wvb, HID * NKV * HD);
  cvt(wo, wob, NH * HD * HID);

  // Q/K/V projections (bf16 out)
  gemm_wmma<true><<<dim3((NH * HD) / 64, TOKS / 128), 256, 0, stream>>>(
      xb, wqb, qb, TOKS, NH * HD, HID);
  gemm_wmma<true><<<dim3((NKV * HD) / 64, TOKS / 128), 256, 0, stream>>>(
      xb, wkb, kb, TOKS, NKV * HD, HID);
  gemm_wmma<true><<<dim3((NKV * HD) / 64, TOKS / 128), 256, 0, stream>>>(
      xb, wvb, vb, TOKS, NKV * HD, HID);

  // RoPE on Q (scaled by 1/sqrt(HD)) and K
  rope_kernel<<<(TOKS * NH * 32) / 256, 256, 0, stream>>>(qb, NH, 0.125f);
  rope_kernel<<<(TOKS * NKV * 32) / 256, 256, 0, stream>>>(kb, NKV, 1.0f);

  // Attention: grid (seq tiles, q heads, batch)
  attn_wmma<<<dim3(SEQ / 128, NH, BSZ), 256, 0, stream>>>(qb, kb, vb, ab);

  // Output projection (fp32 out)
  gemm_wmma<false><<<dim3(HID / 64, TOKS / 128), 256, 0, stream>>>(
      ab, wob, out, TOKS, HID, NH * HD);
}
